// TreeLSTM_33638183863177
// MI455X (gfx1250) — matrix operand order repaired and checked by reading, not compiled, not run on hardware
//
#include <hip/hip_runtime.h>

// ---------------------------------------------------------------------------
// TreeLSTM on MI455X (gfx1250): f16 WMMA GEMMs fed from LDS-staged weights
// (async global->LDS copies), f16 h storage, f32 c storage.
// ---------------------------------------------------------------------------

typedef __attribute__((ext_vector_type(16))) _Float16 v16h;
typedef __attribute__((ext_vector_type(8)))  float    v8f;

#define NL         262144     // leaves
#define HDIM       128
#define TOTAL_ROWS 522240     // sum of all level sizes

union AFrag {
    v16h v;
    _Float16 h[16];
    unsigned int u[8];
};
union BFrag {
    v16h v;
    uint4 q[2];
};

__device__ __forceinline__ float sigmoidf_(float x) {
    return 1.0f / (1.0f + __expf(-x));
}

// Async copy of one 16-byte chunk: global -> LDS (CDNA5 ASYNCcnt path).
__device__ __forceinline__ void async_copy16(unsigned lds_byte_off,
                                             const _Float16* gsrc) {
    asm volatile("global_load_async_to_lds_b128 %0, %1, off"
                 :: "v"(lds_byte_off), "v"(gsrc)
                 : "memory");
}

__device__ __forceinline__ void async_wait0() {
    asm volatile("s_wait_asynccnt 0x0" ::: "memory");
}

// Load a B fragment from LDS: 16 contiguous halfs per lane (2x ds_load_b128).
__device__ __forceinline__ v16h load_bfrag_lds(const _Float16* lds, int frag,
                                               int lane) {
    BFrag b;
    const uint4* p = (const uint4*)(lds + (size_t)frag * 512 + lane * 16);
    b.q[0] = p[0];
    b.q[1] = p[1];
    return b.v;
}

// K-pair offset within a 16x32 f16 A fragment (contiguous order, matches B):
// lanes 0-15: V0..7 -> K{0,2,...,14}; lanes 16-31: +16.
__device__ __forceinline__ int a_koff(int p, int hi) {
    return 2 * p + 16 * hi;
}

// ---------------------------------------------------------------------------
// Weight packing kernels (f32 -> f16, WMMA B-fragment order).
// ---------------------------------------------------------------------------

// B_leaf: [K=128, N=384], B[k][n] = W_iou_w[n*128 + k]. 24 tiles x 4 ksteps.
__global__ __launch_bounds__(256) void pack_b_leaf(const float* __restrict__ W,
                                                   _Float16* __restrict__ out) {
    int idx = blockIdx.x * 256 + threadIdx.x;          // 24*4*512 = 49152
    int hh   = idx & 15;
    int lane = (idx >> 4) & 31;
    int fs   = idx >> 9;
    int ks   = fs & 3;
    int tn   = fs >> 2;
    int k = ks * 32 + 16 * (lane >> 4) + hh;           // 0..127
    int n = tn * 16 + (lane & 15);                     // 0..383
    out[idx] = (_Float16)W[n * 128 + k];
}

// B_int: [K=256, N=640], cols 0..255 = U_f^T, cols 256..639 = U_iou^T.
__global__ __launch_bounds__(256) void pack_b_int(const float* __restrict__ Uf,
                                                  const float* __restrict__ Uiou,
                                                  _Float16* __restrict__ out) {
    int idx = blockIdx.x * 256 + threadIdx.x;          // 40*8*512 = 163840
    int hh   = idx & 15;
    int lane = (idx >> 4) & 31;
    int fs   = idx >> 9;
    int ks   = fs & 7;
    int tn   = fs >> 3;
    int k = ks * 32 + 16 * (lane >> 4) + hh;           // 0..255
    int n = tn * 16 + (lane & 15);                     // 0..639
    float v = (n < 256) ? Uf[n * 256 + k] : Uiou[(n - 256) * 256 + k];
    out[idx] = (_Float16)v;
}

// B_lin: [K=128, N=16] (cols 5..15 zero padded).
__global__ __launch_bounds__(256) void pack_b_lin(const float* __restrict__ L,
                                                  _Float16* __restrict__ out) {
    int idx = blockIdx.x * 256 + threadIdx.x;          // 1*4*512 = 2048
    int hh   = idx & 15;
    int lane = (idx >> 4) & 31;
    int ks   = idx >> 9;                               // tn == 0
    int k = ks * 32 + 16 * (lane >> 4) + hh;           // 0..127
    int n = lane & 15;                                 // 0..15
    float v = (n < 5) ? L[n * 128 + k] : 0.0f;
    out[idx] = (_Float16)v;
}

// ---------------------------------------------------------------------------
// Leaf frontier: x = emb[wordid]; iou = x @ W_iou^T + b; c = sig(i)*tanh(u);
// h = sig(o)*tanh(c). 8 waves/block, 16 rows/wave, K=128 (4 ksteps).
// Entire B_leaf (96 KB) staged in LDS via async copies.
// ---------------------------------------------------------------------------
__global__ __launch_bounds__(256) void leaf_kernel(
    const int* __restrict__ wordid,
    const float* __restrict__ emb,
    const _Float16* __restrict__ Bleaf,
    const float* __restrict__ Wiou_b,
    _Float16* __restrict__ h_out,   // [NL,128] f16
    float* __restrict__ c_out)      // [NL,128] f32
{
    extern __shared__ _Float16 smem[];   // 96 KB = 96 frags * 1 KB
    const int tid  = threadIdx.x;
    const int lane = tid & 31;
    const int wave = tid >> 5;
    const int m    = lane & 15;
    const int hi   = lane >> 4;
    const int row_base = (blockIdx.x * 8 + wave) * 16;

    // Stage all of B_leaf into LDS (6144 x 16B, async).
    for (int i = tid; i < 6144; i += 256) {
        unsigned dst = (unsigned)(unsigned long long)&smem[(size_t)i * 8];
        async_copy16(dst, Bleaf + (size_t)i * 8);
    }

    // Gather + convert 16x128 A tile from embedding rows (per-lane row m).
    const int wid = wordid[row_base + m];
    const float* xrow = emb + (size_t)wid * HDIM;
    AFrag a[4];
#pragma unroll
    for (int ks = 0; ks < 4; ++ks) {
#pragma unroll
        for (int p = 0; p < 8; ++p) {
            int ko = a_koff(p, hi);
            float2 f = *(const float2*)(xrow + ks * 32 + ko);
            a[ks].h[2 * p]     = (_Float16)f.x;
            a[ks].h[2 * p + 1] = (_Float16)f.y;
        }
    }

    async_wait0();
    __syncthreads();

    const int nn = m;   // D-layout column within a 16-col tile
#pragma unroll 1
    for (int t = 0; t < 8; ++t) {
        float bi = Wiou_b[t * 16 + nn];
        float bo = Wiou_b[128 + t * 16 + nn];
        float bu = Wiou_b[256 + t * 16 + nn];
        v8f acci, acco, accu;
#pragma unroll
        for (int r = 0; r < 8; ++r) { acci[r] = bi; acco[r] = bo; accu[r] = bu; }

#pragma unroll
        for (int ks = 0; ks < 4; ++ks) {
            v16h b0 = load_bfrag_lds(smem, (t) * 4 + ks, lane);
            v16h b1 = load_bfrag_lds(smem, (t + 8) * 4 + ks, lane);
            v16h b2 = load_bfrag_lds(smem, (t + 16) * 4 + ks, lane);
            acci = __builtin_amdgcn_wmma_f32_16x16x32_f16(
                false, a[ks].v, false, b0, (short)0, acci, false, false);
            acco = __builtin_amdgcn_wmma_f32_16x16x32_f16(
                false, a[ks].v, false, b1, (short)0, acco, false, false);
            accu = __builtin_amdgcn_wmma_f32_16x16x32_f16(
                false, a[ks].v, false, b2, (short)0, accu, false, false);
        }

#pragma unroll
        for (int r = 0; r < 8; ++r) {
            int drow = row_base + r + 8 * hi;
            int col  = t * 16 + nn;
            float cc = sigmoidf_(acci[r]) * tanhf(accu[r]);
            float hh = sigmoidf_(acco[r]) * tanhf(cc);
            c_out[(size_t)drow * HDIM + col] = cc;
            h_out[(size_t)drow * HDIM + col] = (_Float16)hh;
        }
    }
}

// ---------------------------------------------------------------------------
// Internal frontier: h_cat[n,256] == h_prev[2n,128] flat. Fused GEMM
// [n,256] x [256,640] (f0|f1|i|o|u), forget-gated child c sum, gates.
// blockIdx.y splits the 8 h-tiles in two halves -> only 20 of 40 B tiles
// (160 KB) staged per block -> 2 blocks/WGP.
// ---------------------------------------------------------------------------
__global__ __launch_bounds__(256) void level_kernel(
    const _Float16* __restrict__ h_prev,  // [2n,128] f16 == [n,256]
    const float* __restrict__ c_prev,     // [2n,128] f32
    const _Float16* __restrict__ Bint,
    const float* __restrict__ Uf_b,       // [256]
    const float* __restrict__ Uiou_b,     // [384]
    _Float16* __restrict__ h_out,         // [n,128] f16
    float* __restrict__ c_out)            // [n,128] f32
{
    extern __shared__ _Float16 smem[];   // 160 KB = 160 frags * 1 KB
    const int tid  = threadIdx.x;
    const int lane = tid & 31;
    const int wave = tid >> 5;
    const int m    = lane & 15;
    const int hi   = lane >> 4;
    const int th   = blockIdx.y;              // which half of the t range
    const int row_base = (blockIdx.x * 8 + wave) * 16;

    // Stage the 20 needed B tiles (gate g in 0..4, local tile tl in 0..3):
    // LDS frag fl = (g*4+tl)*8+ks  <-  global frag ((g*8 + th*4 + tl)*8+ks).
    for (int i = tid; i < 10240; i += 256) {   // 160 frags * 64 x 16B
        int frag = i >> 6;
        int part = i & 63;
        int ks = frag & 7;
        int gt = frag >> 3;          // g*4 + tl
        int g  = gt >> 2;
        int tl = gt & 3;
        int tn = g * 8 + th * 4 + tl;
        const _Float16* src = Bint + ((size_t)(tn * 8 + ks) * 512) + part * 8;
        unsigned dst = (unsigned)(unsigned long long)
            &smem[(size_t)frag * 512 + part * 8];
        async_copy16(dst, src);
    }

    // A fragments: 16 rows x 256 K of h_cat, kept in registers (reused 20x).
    const _Float16* arow = h_prev + (size_t)(row_base + m) * 256;
    AFrag a[8];
#pragma unroll
    for (int ks = 0; ks < 8; ++ks) {
#pragma unroll
        for (int p = 0; p < 8; ++p) {
            int ko = a_koff(p, hi);
            a[ks].u[p] = *(const unsigned int*)(arow + ks * 32 + ko);
        }
    }

    async_wait0();
    __syncthreads();

    const int nn = m;
#pragma unroll 1
    for (int tl = 0; tl < 4; ++tl) {
        const int t = th * 4 + tl;
        float bf0 = Uf_b[t * 16 + nn];
        float bf1 = Uf_b[128 + t * 16 + nn];
        float bi  = Uiou_b[t * 16 + nn];
        float bo  = Uiou_b[128 + t * 16 + nn];
        float bu  = Uiou_b[256 + t * 16 + nn];
        v8f accf0, accf1, acci, acco, accu;
#pragma unroll
        for (int r = 0; r < 8; ++r) {
            accf0[r] = bf0; accf1[r] = bf1;
            acci[r] = bi; acco[r] = bo; accu[r] = bu;
        }

#pragma unroll
        for (int ks = 0; ks < 8; ++ks) {
            v16h b0 = load_bfrag_lds(smem, (0 * 4 + tl) * 8 + ks, lane);
            v16h b1 = load_bfrag_lds(smem, (1 * 4 + tl) * 8 + ks, lane);
            v16h b2 = load_bfrag_lds(smem, (2 * 4 + tl) * 8 + ks, lane);
            v16h b3 = load_bfrag_lds(smem, (3 * 4 + tl) * 8 + ks, lane);
            v16h b4 = load_bfrag_lds(smem, (4 * 4 + tl) * 8 + ks, lane);
            accf0 = __builtin_amdgcn_wmma_f32_16x16x32_f16(
                false, a[ks].v, false, b0, (short)0, accf0, false, false);
            accf1 = __builtin_amdgcn_wmma_f32_16x16x32_f16(
                false, a[ks].v, false, b1, (short)0, accf1, false, false);
            acci = __builtin_amdgcn_wmma_f32_16x16x32_f16(
                false, a[ks].v, false, b2, (short)0, acci, false, false);
            acco = __builtin_amdgcn_wmma_f32_16x16x32_f16(
                false, a[ks].v, false, b3, (short)0, acco, false, false);
            accu = __builtin_amdgcn_wmma_f32_16x16x32_f16(
                false, a[ks].v, false, b4, (short)0, accu, false, false);
        }

#pragma unroll
        for (int r = 0; r < 8; ++r) {
            int drow = row_base + r + 8 * hi;    // node index j
            int col  = t * 16 + nn;              // h dim 0..127
            float f0 = sigmoidf_(accf0[r]);
            float f1 = sigmoidf_(accf1[r]);
            float c0 = c_prev[(size_t)(2 * drow) * HDIM + col];
            float c1 = c_prev[(size_t)(2 * drow + 1) * HDIM + col];
            float cf = f0 * c0 + f1 * c1;
            float cc = sigmoidf_(acci[r]) * tanhf(accu[r]) + cf;
            float hh = sigmoidf_(acco[r]) * tanhf(cc);
            c_out[(size_t)drow * HDIM + col] = cc;
            h_out[(size_t)drow * HDIM + col] = (_Float16)hh;
        }
    }
}

// ---------------------------------------------------------------------------
// Final linear: logits = h_all @ lin_w^T + lin_b, N padded 5 -> 16.
// ---------------------------------------------------------------------------
__global__ __launch_bounds__(256) void linear_kernel(
    const _Float16* __restrict__ h_all,  // [522240,128] f16
    const _Float16* __restrict__ Blin,
    const float* __restrict__ lin_b,     // [5]
    float* __restrict__ out)             // [522240,5] f32
{
    extern __shared__ _Float16 smem[];   // 4 KB = 4 frags
    const int tid  = threadIdx.x;
    const int lane = tid & 31;
    const int wave = tid >> 5;
    const int m    = lane & 15;
    const int hi   = lane >> 4;
    const int row_base = (blockIdx.x * 8 + wave) * 16;

    {   // 256 x 16B staging, one chunk per thread
        unsigned dst = (unsigned)(unsigned long long)&smem[(size_t)tid * 8];
        async_copy16(dst, Blin + (size_t)tid * 8);
    }

    const _Float16* arow = h_all + (size_t)(row_base + m) * HDIM;
    const int nn = m;
    v8f acc;
    float b = (nn < 5) ? lin_b[nn] : 0.0f;
#pragma unroll
    for (int r = 0; r < 8; ++r) acc[r] = b;

    AFrag a[4];
#pragma unroll
    for (int ks = 0; ks < 4; ++ks) {
#pragma unroll
        for (int p = 0; p < 8; ++p) {
            int ko = a_koff(p, hi);
            a[ks].u[p] = *(const unsigned int*)(arow + ks * 32 + ko);
        }
    }

    async_wait0();
    __syncthreads();

#pragma unroll
    for (int ks = 0; ks < 4; ++ks) {
        acc = __builtin_amdgcn_wmma_f32_16x16x32_f16(
            false, a[ks].v, false, load_bfrag_lds(smem, ks, lane),
            (short)0, acc, false, false);
    }

    if (nn < 5) {
#pragma unroll
        for (int r = 0; r < 8; ++r) {
            int drow = row_base + r + 8 * hi;
            out[(size_t)drow * 5 + nn] = acc[r];
        }
    }
}

// ---------------------------------------------------------------------------
// Host launch
// ---------------------------------------------------------------------------

// Workspace layout (bytes, all 256-aligned)
#define H_ALL_OFF   0UL
#define H_ALL_BYTES (522240UL * 128 * 2)          // 133,693,440
#define C_A_OFF     (H_ALL_OFF + H_ALL_BYTES)
#define C_A_BYTES   (262144UL * 128 * 4)          // 134,217,728
#define C_B_OFF     (C_A_OFF + C_A_BYTES)
#define C_B_BYTES   (131072UL * 128 * 4)          // 67,108,864
#define BLEAF_OFF   (C_B_OFF + C_B_BYTES)
#define BLEAF_BYTES (24UL * 4 * 512 * 2)          // 98,304
#define BINT_OFF    (BLEAF_OFF + BLEAF_BYTES)
#define BINT_BYTES  (40UL * 8 * 512 * 2)          // 327,680
#define BLIN_OFF    (BINT_OFF + BINT_BYTES)

extern "C" void kernel_launch(void* const* d_in, const int* in_sizes, int n_in,
                              void* d_out, int out_size, void* d_ws, size_t ws_size,
                              hipStream_t stream) {
    const int*   wordid = (const int*)d_in[0];
    const float* emb    = (const float*)d_in[1];
    const float* Wiou_w = (const float*)d_in[2];
    const float* Wiou_b = (const float*)d_in[3];
    const float* Uiou_w = (const float*)d_in[4];
    const float* Uiou_b = (const float*)d_in[5];
    const float* Uf_w   = (const float*)d_in[6];
    const float* Uf_b   = (const float*)d_in[7];
    const float* lin_w  = (const float*)d_in[8];
    const float* lin_b  = (const float*)d_in[9];
    float* out = (float*)d_out;

    char* ws = (char*)d_ws;
    _Float16* h_all = (_Float16*)(ws + H_ALL_OFF);
    float*    c_a   = (float*)(ws + C_A_OFF);
    float*    c_b   = (float*)(ws + C_B_OFF);
    _Float16* Bleaf = (_Float16*)(ws + BLEAF_OFF);
    _Float16* Bint  = (_Float16*)(ws + BINT_OFF);
    _Float16* Blin  = (_Float16*)(ws + BLIN_OFF);

    // 1) Pack weights into f16 WMMA-fragment order (tiny, L2-resident).
    pack_b_leaf<<<192, 256, 0, stream>>>(Wiou_w, Bleaf);
    pack_b_int<<<640, 256, 0, stream>>>(Uf_w, Uiou_w, Bint);
    pack_b_lin<<<8, 256, 0, stream>>>(lin_w, Blin);

    // 2) Leaf frontier: 262144 nodes, 128 nodes/block, 96 KB LDS.
    leaf_kernel<<<NL / 128, 256, 98304, stream>>>(wordid, emb, Bleaf, Wiou_b,
                                                  h_all, c_a);

    // 3) Internal frontiers (ping-pong c buffers, h appended level-major).
    size_t h_off = 0;
    size_t rows_prev = NL;
    float* cp = c_a;
    float* cn = c_b;
    for (int lvl = 1; lvl < 8; ++lvl) {
        int n = NL >> lvl;
        _Float16* hp = h_all + h_off * HDIM;
        _Float16* ho = h_all + (h_off + rows_prev) * HDIM;
        level_kernel<<<dim3(n / 128, 2), 256, 163840, stream>>>(
            hp, cp, Bint, Uf_b, Uiou_b, ho, cn);
        h_off += rows_prev;
        rows_prev = (size_t)n;
        float* t = cp; cp = cn; cn = t;
    }

    // 4) Final linear over all 522240 rows, 4 KB LDS.
    linear_kernel<<<TOTAL_ROWS / 128, 256, 4096, stream>>>(h_all, Blin, lin_b,
                                                           out);
}
